// EdgePool_24867860644344
// MI455X (gfx1250) — matrix operand order repaired and checked by reading, not compiled, not run on hardware
//
#include <hip/hip_runtime.h>
#include <hip/hip_bf16.h>

typedef __attribute__((ext_vector_type(16))) _Float16 v16h;
typedef __attribute__((ext_vector_type(8)))  _Float16 h8;
typedef __attribute__((ext_vector_type(4)))  _Float16 h4;
typedef __attribute__((ext_vector_type(8)))  float    v8f;

// ---------------- helpers ----------------
__device__ __forceinline__ unsigned f2o(float f) {
  unsigned u = __float_as_uint(f);
  return (u & 0x80000000u) ? ~u : (u | 0x80000000u);
}
__device__ __forceinline__ float o2f(unsigned u) {
  return (u & 0x80000000u) ? __uint_as_float(u & 0x7FFFFFFFu) : __uint_as_float(~u);
}

__global__ void fill_u32_kernel(unsigned* p, unsigned v, long long n) {
  long long i = blockIdx.x * (long long)blockDim.x + threadIdx.x;
  if (i < n) p[i] = v;
}
__global__ void fill_u64_kernel(unsigned long long* p, unsigned long long v, long long n) {
  long long i = blockIdx.x * (long long)blockDim.x + threadIdx.x;
  if (i < n) p[i] = v;
}

// ---------------- graph conv: scatter mean + WMMA GEMM ----------------
__global__ void degree_kernel(const int* __restrict__ dst, float* __restrict__ cnt, int E, int N) {
  int e = blockIdx.x * blockDim.x + threadIdx.x;
  if (e >= E) return;
  int d = dst[e];
  if ((unsigned)d < (unsigned)N) atomicAdd(&cnt[d], 1.0f);
}

template <int F>
__global__ void scatter_sum_kernel(const float* __restrict__ x, const int* __restrict__ src,
                                   const int* __restrict__ dst, float* __restrict__ sum,
                                   int E, int N) {
  const int CH = F / 4;
  long long tid = blockIdx.x * (long long)blockDim.x + threadIdx.x;
  long long e = tid / CH;
  int c = (int)(tid % CH);
  if (e >= E) return;
  int s = src[e], d = dst[e];
  if ((unsigned)s >= (unsigned)N || (unsigned)d >= (unsigned)N) return;  // padded edges
  float4 v = ((const float4*)(x + (size_t)s * F))[c];
  float* b = sum + (size_t)d * F + (size_t)c * 4;
  atomicAdd(b + 0, v.x); atomicAdd(b + 1, v.y);
  atomicAdd(b + 2, v.z); atomicAdd(b + 3, v.w);
}

// out[N,64] = relu( (sum/cnt) @ w_rel^T + b_rel + x @ w_root^T )
// block = 128 thr (4 waves). Block owns 16 rows; wave w owns cols [16w,16w+16).
// Activations staged once per block into LDS as f16 (kills the 4x re-read),
// weights staged once as f16; all fragment reads are ds_load_b128-class.
template <int F>
__global__ void __launch_bounds__(128)
gconv_wmma_kernel(const float* __restrict__ xsum, const float* __restrict__ cnt,
                  const float* __restrict__ x,
                  const float* __restrict__ w_rel, const float* __restrict__ b_rel,
                  const float* __restrict__ w_root,
                  float* __restrict__ out, int N) {
  constexpr int H = 64;
  __shared__ __align__(32) _Float16 sWrel[H * F];
  __shared__ __align__(32) _Float16 sWroot[H * F];
  __shared__ __align__(32) _Float16 sAm[16 * F];
  __shared__ __align__(32) _Float16 sAx[16 * F];

  const int tid = threadIdx.x;
  const int rowbase = blockIdx.x * 16;

  // stage weights: float4 global loads -> packed h4 LDS stores
#pragma unroll
  for (int i = tid * 4; i < H * F; i += 128 * 4) {
    float4 a = *(const float4*)(w_rel + i);
    float4 b = *(const float4*)(w_root + i);
    h4 ha = {(_Float16)a.x, (_Float16)a.y, (_Float16)a.z, (_Float16)a.w};
    h4 hb = {(_Float16)b.x, (_Float16)b.y, (_Float16)b.z, (_Float16)b.w};
    *(h4*)(sWrel + i) = ha;
    *(h4*)(sWroot + i) = hb;
  }
  // stage A tile: 16 rows of mean (scaled) and x
  {
    constexpr int CH = F / 4;
#pragma unroll
    for (int i = tid; i < 16 * CH; i += 128) {
      int r = i / CH, c = i % CH;
      int gr = rowbase + r;
      int grc = (gr < N) ? gr : (N - 1);  // clamp; stores are row-guarded later
      float inv = 1.0f / fmaxf(cnt[grc], 1.0f);
      float4 aS = ((const float4*)(xsum + (size_t)grc * F))[c];
      float4 aX = ((const float4*)(x + (size_t)grc * F))[c];
      h4 hm = {(_Float16)(aS.x * inv), (_Float16)(aS.y * inv),
               (_Float16)(aS.z * inv), (_Float16)(aS.w * inv)};
      h4 hx = {(_Float16)aX.x, (_Float16)aX.y, (_Float16)aX.z, (_Float16)aX.w};
      *(h4*)(sAm + r * F + c * 4) = hm;
      *(h4*)(sAx + r * F + c * 4) = hx;
    }
  }
  __syncthreads();

  const int lane = tid & 31;
  const int wave = tid >> 5;
  const int hi = lane >> 4;                  // lane group 0/1
  const int mrow = lane & 15;
  const int ncol = wave * 16 + (lane & 15);  // output column (0..63)

  v8f acc = {};
#pragma unroll
  for (int kb = 0; kb < F; kb += 32) {
    // A layout (matches B): elem i -> K=kb+hi*16+i
    h8 m0 = *(const h8*)(sAm + mrow * F + kb + hi * 16);
    h8 m1 = *(const h8*)(sAm + mrow * F + kb + hi * 16 + 8);
    h8 x0 = *(const h8*)(sAx + mrow * F + kb + hi * 16);
    h8 x1 = *(const h8*)(sAx + mrow * F + kb + hi * 16 + 8);
    v16h aM, aX;
#pragma unroll
    for (int i = 0; i < 8; ++i) {
      aM[i] = m0[i]; aM[i + 8] = m1[i];
      aX[i] = x0[i]; aX[i + 8] = x1[i];
    }
    // B 32x16 layout (contiguous order, same as A): K = hi*16 + i, N = lane%16 (contiguous 16 halfs)
    v16h bR = *(const v16h*)(sWrel  + (size_t)ncol * F + kb + hi * 16);
    v16h bT = *(const v16h*)(sWroot + (size_t)ncol * F + kb + hi * 16);
    acc = __builtin_amdgcn_wmma_f32_16x16x32_f16(false, aM, false, bR, (short)0, acc, false, false);
    acc = __builtin_amdgcn_wmma_f32_16x16x32_f16(false, aX, false, bT, (short)0, acc, false, false);
  }
  float bias = b_rel[ncol];
#pragma unroll
  for (int v = 0; v < 8; ++v) {
    int r = rowbase + v + 8 * hi;  // C/D layout: M = v + 8*(lane/16)
    if (r < N) out[(size_t)r * H + ncol] = fmaxf(acc[v] + bias, 0.0f);
  }
}

// ---------------- global mean pool (LDS ds_add_f32 staging) ----------------
__global__ void gmp_accum_kernel(const float* __restrict__ h, const int* __restrict__ batch,
                                 float* __restrict__ gsum, float* __restrict__ gcnt, int N, int G) {
  __shared__ float ls[9 * 64];
  __shared__ float lc[9];
  int t = threadIdx.x;
  for (int i = t; i < 9 * 64; i += 256) ls[i] = 0.0f;
  if (t < 9) lc[t] = 0.0f;
  __syncthreads();
  int node = blockIdx.x * 64 + (t >> 2);
  if (node < N) {
    int b = batch[node];
    if ((unsigned)b >= (unsigned)G) b = 8;   // padded nodes -> dropped row
    int f0 = (t & 3) * 16;
    if ((t & 3) == 0) atomicAdd(&lc[b], 1.0f);
    const float* hp = h + (size_t)node * 64 + f0;
    for (int i = 0; i < 16; ++i) atomicAdd(&ls[b * 64 + f0 + i], hp[i]);
  }
  __syncthreads();
  for (int i = t; i < G * 64; i += 256) atomicAdd(&gsum[i], ls[i]);
  if (t < G) atomicAdd(&gcnt[t], lc[t]);
}

__global__ void gmp_finalize_kernel(const float* __restrict__ gsum, const float* __restrict__ gcnt,
                                    float* __restrict__ xs, int G, int slot) {
  int i = blockIdx.x * blockDim.x + threadIdx.x;
  if (i >= G * 64) return;
  int g = i >> 6, f = i & 63;
  xs[(size_t)g * 256 + slot * 64 + f] = gsum[i] / fmaxf(gcnt[g], 1.0f);
}

// ---------------- edge pooling ----------------
__global__ void edge_raw_kernel(const float* __restrict__ h, const int* __restrict__ src,
                                const int* __restrict__ dst, const float* __restrict__ pw,
                                const float* __restrict__ pb, float* __restrict__ raw,
                                int E, int N) {
  int wid = blockIdx.x * (blockDim.x >> 5) + (threadIdx.x >> 5);
  int lane = threadIdx.x & 31;
  if (wid >= E) return;
  int s = src[wid], d = dst[wid];
  float p = 0.0f;
  if ((unsigned)s < (unsigned)N && (unsigned)d < (unsigned)N) {
    const float* hs = h + (size_t)s * 64;
    const float* hd = h + (size_t)d * 64;
    p = hs[lane] * pw[lane] + hs[lane + 32] * pw[lane + 32]
      + hd[lane] * pw[64 + lane] + hd[lane + 32] * pw[96 + lane];
  }
  for (int off = 16; off > 0; off >>= 1) p += __shfl_xor(p, off, 32);
  if (lane == 0) raw[wid] = p + pb[0];
}

__global__ void segmax_kernel(const float* __restrict__ raw, const int* __restrict__ dst,
                              unsigned* __restrict__ m, int E, int N) {
  int e = blockIdx.x * blockDim.x + threadIdx.x;
  if (e >= E) return;
  int d = dst[e];
  if ((unsigned)d < (unsigned)N) atomicMax(&m[d], f2o(raw[e]));
}

__global__ void edge_exp_kernel(const float* __restrict__ raw, const unsigned* __restrict__ m,
                                const int* __restrict__ dst, float* __restrict__ ex,
                                float* __restrict__ den, int E, int N) {
  int e = blockIdx.x * blockDim.x + threadIdx.x;
  if (e >= E) return;
  int d = dst[e];
  float v = 0.0f;
  if ((unsigned)d < (unsigned)N) {
    v = __expf(raw[e] - o2f(m[d]));
    atomicAdd(&den[d], v);
  }
  ex[e] = v;
}

__global__ void edge_score_kernel(const float* __restrict__ ex, const float* __restrict__ den,
                                  const int* __restrict__ dst, float* __restrict__ score,
                                  int E, int N) {
  int e = blockIdx.x * blockDim.x + threadIdx.x;
  if (e >= E) return;
  int d = dst[e];
  float s = 0.5f;
  if ((unsigned)d < (unsigned)N) s = ex[e] / fmaxf(den[d], 1e-30f) + 0.5f;
  score[e] = s;
}

__global__ void build_keys_kernel(const float* __restrict__ score, unsigned long long* __restrict__ keys,
                                  int E, int M) {
  int i = blockIdx.x * blockDim.x + threadIdx.x;
  if (i >= M) return;
  if (i < E) keys[i] = ((unsigned long long)(~f2o(score[i])) << 32) | (unsigned)i;
  else       keys[i] = ~0ull;
}

__global__ void bitonic_step_kernel(unsigned long long* __restrict__ keys, int j, int k, int n) {
  int i = blockIdx.x * blockDim.x + threadIdx.x;
  if (i >= n) return;
  int ixj = i ^ j;
  if (ixj > i) {
    unsigned long long a = keys[i], b = keys[ixj];
    bool up = ((i & k) == 0);
    if ((a > b) == up && a != b) { keys[i] = b; keys[ixj] = a; }
  }
}

__global__ void greedy_kernel(const unsigned long long* __restrict__ keys,
                              const float* __restrict__ score,
                              const int* __restrict__ src, const int* __restrict__ dst,
                              int* __restrict__ cluster, int* __restrict__ rem,
                              float* __restrict__ cs, int E, int N) {
  if (blockIdx.x != 0 || threadIdx.x != 0) return;
  for (int n = 0; n < N; ++n) { cluster[n] = 0; rem[n] = 1; cs[n] = 1.0f; }
  int cnt = 0;
  for (int e = 0; e < E; ++e) {
    int idx = (int)(keys[e] & 0xFFFFFFFFu);
    int s = src[idx], d = dst[idx];
    if (rem[s] && rem[d]) {
      cluster[s] = cnt; cluster[d] = cnt;
      rem[s] = 0; rem[d] = 0;
      cs[cnt] = score[idx];
      ++cnt;
    }
  }
  for (int n = 0; n < N; ++n) {
    if (rem[n]) { cluster[n] = cnt; cs[cnt] = 1.0f; ++cnt; }
  }
}

__global__ void newx_accum_kernel(const float* __restrict__ h, const int* __restrict__ cluster,
                                  float* __restrict__ nx, int N) {
  long long tid = blockIdx.x * (long long)blockDim.x + threadIdx.x;
  long long n = tid >> 4;
  int q = (int)(tid & 15);
  if (n >= N) return;
  int c = cluster[n];
  float4 v = ((const float4*)(h + (size_t)n * 64))[q];
  float* b = nx + (size_t)c * 64 + (size_t)q * 4;
  atomicAdd(b + 0, v.x); atomicAdd(b + 1, v.y);
  atomicAdd(b + 2, v.z); atomicAdd(b + 3, v.w);
}

__global__ void newx_scale_kernel(float* __restrict__ nx, const float* __restrict__ cs, int N) {
  long long i = blockIdx.x * (long long)blockDim.x + threadIdx.x;
  if (i < (long long)N * 64) nx[i] *= cs[i >> 6];
}

__global__ void set_batch_kernel(const int* __restrict__ cluster, const int* __restrict__ batch,
                                 int* __restrict__ nbatch, int N) {
  int n = blockIdx.x * blockDim.x + threadIdx.x;
  if (n < N) nbatch[cluster[n]] = batch[n];  // same graph per cluster: benign race
}

__global__ void remap_kernel(const int* __restrict__ cluster, const int* __restrict__ src,
                             const int* __restrict__ dst, int* __restrict__ ns,
                             int* __restrict__ nd, int E) {
  int e = blockIdx.x * blockDim.x + threadIdx.x;
  if (e >= E) return;
  ns[e] = cluster[src[e]];
  nd[e] = cluster[dst[e]];
}

__global__ void dedup_insert_kernel(const int* __restrict__ ns, const int* __restrict__ nd,
                                    unsigned long long* __restrict__ hk, unsigned* __restrict__ hv,
                                    int E, int HSZ) {
  int e = blockIdx.x * blockDim.x + threadIdx.x;
  if (e >= E) return;
  unsigned long long key = ((unsigned long long)(unsigned)ns[e] << 32) | (unsigned)nd[e];
  unsigned h = (unsigned)((key * 0x9E3779B97F4A7C15ull) >> 43) & (unsigned)(HSZ - 1);
  for (;;) {
    unsigned long long old = atomicCAS(&hk[h], ~0ull, key);
    if (old == ~0ull || old == key) { atomicMin(&hv[h], (unsigned)e); return; }
    h = (h + 1) & (unsigned)(HSZ - 1);
  }
}

__global__ void dedup_filter_kernel(const int* __restrict__ ns, const int* __restrict__ nd,
                                    const unsigned long long* __restrict__ hk,
                                    const unsigned* __restrict__ hv,
                                    int* __restrict__ ns2, int* __restrict__ nd2,
                                    int E, int N, int HSZ) {
  int e = blockIdx.x * blockDim.x + threadIdx.x;
  if (e >= E) return;
  int s = ns[e], d = nd[e];
  unsigned long long key = ((unsigned long long)(unsigned)s << 32) | (unsigned)d;
  unsigned h = (unsigned)((key * 0x9E3779B97F4A7C15ull) >> 43) & (unsigned)(HSZ - 1);
  while (hk[h] != key) h = (h + 1) & (unsigned)(HSZ - 1);
  bool keep = (hv[h] == (unsigned)e);
  ns2[e] = keep ? s : N;
  nd2[e] = keep ? d : N;
}

// ---------------- MLP head + log_softmax ----------------
__global__ void head_kernel(const float* __restrict__ xs, const float* __restrict__ w1,
                            const float* __restrict__ b1, const float* __restrict__ w2,
                            const float* __restrict__ b2, float* __restrict__ out,
                            int G, int C) {
  __shared__ float z1[8 * 64];
  __shared__ float logit[8 * 16];
  int t = threadIdx.x;
  for (int i = t; i < G * 64; i += 256) {
    int g = i >> 6, j = i & 63;
    float acc = b1[j];
    const float* xg = xs + (size_t)g * 256;
    const float* wj = w1 + (size_t)j * 256;
    for (int k = 0; k < 256; ++k) acc += xg[k] * wj[k];
    z1[i] = fmaxf(acc, 0.0f);
  }
  __syncthreads();
  for (int i = t; i < G * C; i += 256) {
    int g = i / C, c = i % C;
    float acc = b2[c];
    const float* zg = z1 + g * 64;
    const float* wc = w2 + (size_t)c * 64;
    for (int k = 0; k < 64; ++k) acc += zg[k] * wc[k];
    logit[g * 16 + c] = acc;
  }
  __syncthreads();
  if (t < G) {
    float mx = -1e30f;
    for (int c = 0; c < C; ++c) mx = fmaxf(mx, logit[t * 16 + c]);
    float s = 0.0f;
    for (int c = 0; c < C; ++c) s += __expf(logit[t * 16 + c] - mx);
    float lse = mx + __logf(s);
    for (int c = 0; c < C; ++c) out[t * C + c] = logit[t * 16 + c] - lse;
  }
}

// ---------------- host orchestration ----------------
extern "C" void kernel_launch(void* const* d_in, const int* in_sizes, int n_in,
                              void* d_out, int out_size, void* d_ws, size_t ws_size,
                              hipStream_t stream) {
  (void)n_in; (void)ws_size;
  const float* x       = (const float*)d_in[0];
  const int*   ei      = (const int*)d_in[1];
  const int*   batch   = (const int*)d_in[2];
  const float* w_rel1  = (const float*)d_in[3];
  const float* b_rel1  = (const float*)d_in[4];
  const float* w_root1 = (const float*)d_in[5];
  const float* w_rel2  = (const float*)d_in[6];
  const float* b_rel2  = (const float*)d_in[7];
  const float* w_root2 = (const float*)d_in[8];
  const float* w_rel3  = (const float*)d_in[9];
  const float* b_rel3  = (const float*)d_in[10];
  const float* w_root3 = (const float*)d_in[11];
  const float* w_rel4  = (const float*)d_in[12];
  const float* b_rel4  = (const float*)d_in[13];
  const float* w_root4 = (const float*)d_in[14];
  const float* pool_w  = (const float*)d_in[15];
  const float* pool_b  = (const float*)d_in[16];
  const float* lin1_w  = (const float*)d_in[17];
  const float* lin1_b  = (const float*)d_in[18];
  const float* lin2_w  = (const float*)d_in[19];
  const float* lin2_b  = (const float*)d_in[20];

  const int N = in_sizes[2];
  const int E = in_sizes[1] / 2;
  const int F = in_sizes[0] / N;       // 128
  const int H = 64;
  const int C = in_sizes[20];          // 10
  const int G = out_size / C;          // 8
  const int SORTN = 1 << 20;           // >= E, power of two
  const int HSZ = 1 << 21;             // hash table slots

  const int* src = ei;
  const int* dst = ei + E;

  char* wp = (char*)d_ws;
  auto carve = [&](size_t bytes) -> void* {
    void* p = (void*)wp;
    wp += (bytes + 255) & ~(size_t)255;
    return p;
  };
  float* sum   = (float*)carve((size_t)N * F * 4);
  float* cnt1  = (float*)carve((size_t)N * 4);
  float* cnt2  = (float*)carve((size_t)N * 4);
  float* bufA  = (float*)carve((size_t)N * H * 4);
  float* bufB  = (float*)carve((size_t)N * H * 4);
  float* bufC  = (float*)carve((size_t)N * H * 4);
  float* gsum  = (float*)carve((size_t)G * H * 4);
  float* gcnt  = (float*)carve((size_t)G * 4);
  float* xs    = (float*)carve((size_t)G * 4 * H * 4);
  float* raw   = (float*)carve((size_t)E * 4);
  unsigned* segm = (unsigned*)carve((size_t)N * 4);
  float* exv   = (float*)carve((size_t)E * 4);
  float* den   = (float*)carve((size_t)N * 4);
  float* score = (float*)carve((size_t)E * 4);
  unsigned long long* keys = (unsigned long long*)carve((size_t)SORTN * 8);
  int* cluster = (int*)carve((size_t)N * 4);
  int* remv    = (int*)carve((size_t)N * 4);
  float* cs    = (float*)carve((size_t)N * 4);
  int* nbatch  = (int*)carve((size_t)N * 4);
  int* ns      = (int*)carve((size_t)E * 4);
  int* nd      = (int*)carve((size_t)E * 4);
  int* ns2     = (int*)carve((size_t)E * 4);
  int* nd2     = (int*)carve((size_t)E * 4);
  unsigned long long* hkeys = (unsigned long long*)carve((size_t)HSZ * 8);
  unsigned* hvals = (unsigned*)carve((size_t)HSZ * 4);

  auto nb = [](long long n, int b) { return (unsigned)((n + b - 1) / b); };
  const int gconvBlocks = (N + 15) / 16;

  // ======== layer 1 (F=128) ========
  fill_u32_kernel<<<nb((long long)N * F, 256), 256, 0, stream>>>((unsigned*)sum, 0u, (long long)N * F);
  fill_u32_kernel<<<nb(N, 256), 256, 0, stream>>>((unsigned*)cnt1, 0u, N);
  degree_kernel<<<nb(E, 256), 256, 0, stream>>>(dst, cnt1, E, N);
  scatter_sum_kernel<128><<<nb((long long)E * 32, 256), 256, 0, stream>>>(x, src, dst, sum, E, N);
  gconv_wmma_kernel<128><<<gconvBlocks, 128, 0, stream>>>(sum, cnt1, x, w_rel1, b_rel1, w_root1, bufA, N);
  fill_u32_kernel<<<nb(G * H, 256), 256, 0, stream>>>((unsigned*)gsum, 0u, G * H);
  fill_u32_kernel<<<1, 64, 0, stream>>>((unsigned*)gcnt, 0u, G);
  gmp_accum_kernel<<<(N + 63) / 64, 256, 0, stream>>>(bufA, batch, gsum, gcnt, N, G);
  gmp_finalize_kernel<<<nb(G * H, 256), 256, 0, stream>>>(gsum, gcnt, xs, G, 0);

  // ======== layer 2 (F=64) ========
  fill_u32_kernel<<<nb((long long)N * H, 256), 256, 0, stream>>>((unsigned*)sum, 0u, (long long)N * H);
  scatter_sum_kernel<64><<<nb((long long)E * 16, 256), 256, 0, stream>>>(bufA, src, dst, sum, E, N);
  gconv_wmma_kernel<64><<<gconvBlocks, 128, 0, stream>>>(sum, cnt1, bufA, w_rel2, b_rel2, w_root2, bufB, N);
  fill_u32_kernel<<<nb(G * H, 256), 256, 0, stream>>>((unsigned*)gsum, 0u, G * H);
  fill_u32_kernel<<<1, 64, 0, stream>>>((unsigned*)gcnt, 0u, G);
  gmp_accum_kernel<<<(N + 63) / 64, 256, 0, stream>>>(bufB, batch, gsum, gcnt, N, G);
  gmp_finalize_kernel<<<nb(G * H, 256), 256, 0, stream>>>(gsum, gcnt, xs, G, 1);

  // ======== edge pooling ========
  edge_raw_kernel<<<nb((long long)E * 32, 256), 256, 0, stream>>>(bufB, src, dst, pool_w, pool_b, raw, E, N);
  fill_u32_kernel<<<nb(N, 256), 256, 0, stream>>>(segm, 0x007FFFFFu, N);   // ordered(-inf)
  segmax_kernel<<<nb(E, 256), 256, 0, stream>>>(raw, dst, segm, E, N);
  fill_u32_kernel<<<nb(N, 256), 256, 0, stream>>>((unsigned*)den, 0u, N);
  edge_exp_kernel<<<nb(E, 256), 256, 0, stream>>>(raw, segm, dst, exv, den, E, N);
  edge_score_kernel<<<nb(E, 256), 256, 0, stream>>>(exv, den, dst, score, E, N);
  build_keys_kernel<<<nb(SORTN, 256), 256, 0, stream>>>(score, keys, E, SORTN);
  for (int k = 2; k <= SORTN; k <<= 1)
    for (int j = k >> 1; j > 0; j >>= 1)
      bitonic_step_kernel<<<nb(SORTN, 256), 256, 0, stream>>>(keys, j, k, SORTN);
  greedy_kernel<<<1, 1, 0, stream>>>(keys, score, src, dst, cluster, remv, cs, E, N);
  fill_u32_kernel<<<nb((long long)N * H, 256), 256, 0, stream>>>((unsigned*)bufC, 0u, (long long)N * H);
  newx_accum_kernel<<<nb((long long)N * 16, 256), 256, 0, stream>>>(bufB, cluster, bufC, N);
  newx_scale_kernel<<<nb((long long)N * H, 256), 256, 0, stream>>>(bufC, cs, N);
  fill_u32_kernel<<<nb(N, 256), 256, 0, stream>>>((unsigned*)nbatch, (unsigned)G, N);
  set_batch_kernel<<<nb(N, 256), 256, 0, stream>>>(cluster, batch, nbatch, N);
  remap_kernel<<<nb(E, 256), 256, 0, stream>>>(cluster, src, dst, ns, nd, E);
  fill_u64_kernel<<<nb(HSZ, 256), 256, 0, stream>>>(hkeys, ~0ull, HSZ);
  fill_u32_kernel<<<nb(HSZ, 256), 256, 0, stream>>>(hvals, 0xFFFFFFFFu, HSZ);
  dedup_insert_kernel<<<nb(E, 256), 256, 0, stream>>>(ns, nd, hkeys, hvals, E, HSZ);
  dedup_filter_kernel<<<nb(E, 256), 256, 0, stream>>>(ns, nd, hkeys, hvals, ns2, nd2, E, N, HSZ);

  // ======== layer 3 (pooled, F=64) ========
  fill_u32_kernel<<<nb((long long)N * H, 256), 256, 0, stream>>>((unsigned*)sum, 0u, (long long)N * H);
  fill_u32_kernel<<<nb(N, 256), 256, 0, stream>>>((unsigned*)cnt2, 0u, N);
  degree_kernel<<<nb(E, 256), 256, 0, stream>>>(nd2, cnt2, E, N);
  scatter_sum_kernel<64><<<nb((long long)E * 16, 256), 256, 0, stream>>>(bufC, ns2, nd2, sum, E, N);
  gconv_wmma_kernel<64><<<gconvBlocks, 128, 0, stream>>>(sum, cnt2, bufC, w_rel3, b_rel3, w_root3, bufA, N);
  fill_u32_kernel<<<nb(G * H, 256), 256, 0, stream>>>((unsigned*)gsum, 0u, G * H);
  fill_u32_kernel<<<1, 64, 0, stream>>>((unsigned*)gcnt, 0u, G);
  gmp_accum_kernel<<<(N + 63) / 64, 256, 0, stream>>>(bufA, nbatch, gsum, gcnt, N, G);
  gmp_finalize_kernel<<<nb(G * H, 256), 256, 0, stream>>>(gsum, gcnt, xs, G, 2);

  // ======== layer 4 (pooled, F=64) ========
  fill_u32_kernel<<<nb((long long)N * H, 256), 256, 0, stream>>>((unsigned*)sum, 0u, (long long)N * H);
  scatter_sum_kernel<64><<<nb((long long)E * 16, 256), 256, 0, stream>>>(bufA, ns2, nd2, sum, E, N);
  gconv_wmma_kernel<64><<<gconvBlocks, 128, 0, stream>>>(sum, cnt2, bufA, w_rel4, b_rel4, w_root4, bufB, N);
  fill_u32_kernel<<<nb(G * H, 256), 256, 0, stream>>>((unsigned*)gsum, 0u, G * H);
  fill_u32_kernel<<<1, 64, 0, stream>>>((unsigned*)gcnt, 0u, G);
  gmp_accum_kernel<<<(N + 63) / 64, 256, 0, stream>>>(bufB, nbatch, gsum, gcnt, N, G);
  gmp_finalize_kernel<<<nb(G * H, 256), 256, 0, stream>>>(gsum, gcnt, xs, G, 3);

  // ======== head ========
  head_kernel<<<1, 256, 0, stream>>>(xs, lin1_w, lin1_b, lin2_w, lin2_b, (float*)d_out, G, C);
}
